// MultiHeadAttention_82652350644938
// MI455X (gfx1250) — matrix operand order repaired and checked by reading, not compiled, not run on hardware
//
#include <hip/hip_runtime.h>

// ---------------------------------------------------------------------------
// MI455X (gfx1250) multi-head causal attention, bf16 WMMA pipeline.
// B=4, T=2048, E=1024, H=16, D=64.
// Round 3: flash-attention now uses 4-wave workgroups with K/V tiles staged
// through LDS via CDNA5 async copies (global_load_async_to_lds_b128 +
// s_wait_asynccnt), double buffered.  GEMMs keep the round-2 ping-pong
// pipelining (partial s_wait_loadcnt overlap confirmed in asm).
// ---------------------------------------------------------------------------

#define BB 4
#define TT 2048
#define EE 1024
#define HH 16
#define DD 64

typedef __bf16 bf16_t;
typedef __attribute__((ext_vector_type(16))) __bf16 v16bf;
typedef __attribute__((ext_vector_type(8)))  float  v8f;

// float -> bf16, round to nearest even
__device__ __forceinline__ bf16_t f2bf(float f) {
    unsigned u = __builtin_bit_cast(unsigned, f);
    unsigned r = (u + 0x7FFFu + ((u >> 16) & 1u)) >> 16;
    unsigned short h = (unsigned short)r;
    return __builtin_bit_cast(bf16_t, h);
}
__device__ __forceinline__ unsigned short f2bf_bits(float f) {
    unsigned u = __builtin_bit_cast(unsigned, f);
    return (unsigned short)((u + 0x7FFFu + ((u >> 16) & 1u)) >> 16);
}

// Load a 32-byte (16 x bf16) fragment as two 16B vector loads.
__device__ __forceinline__ v16bf load_frag(const bf16_t* p0, const bf16_t* p1) {
    v16bf r;
    ((uint4*)&r)[0] = *(const uint4*)p0;
    ((uint4*)&r)[1] = *(const uint4*)p1;
    return r;
}

__device__ __forceinline__ v8f wmma_bf16(v16bf a, v16bf b, v8f c) {
    return __builtin_amdgcn_wmma_f32_16x16x32_bf16(
        false, a, false, b, (short)0, c, false, false);
}

// A-fragment (16xK): lane m=l&15, half=l>>4;
//  elems 0..7 -> k = k0+half*8+e ; elems 8..15 -> k = k0+16+half*8+(e-8)
__device__ __forceinline__ v16bf load_a(const bf16_t* row, int k0, int half) {
    return load_frag(row + k0 + half * 8, row + k0 + 16 + half * 8);
}
// B-fragment (Kx16): lane n=l&15, half=l>>4;
//  elems 0..7 -> k = k0+half*8+e ; elems 8..15 -> k = k0+16+half*8+(e-8)
__device__ __forceinline__ v16bf load_b(const bf16_t* col, int k0, int half) {
    return load_frag(col + k0 + half * 8, col + k0 + 16 + half * 8);
}

// ---- CDNA5 async global->LDS copy (16 bytes per active lane) --------------
__device__ __forceinline__ void async_ld16(unsigned lds_off, const void* g) {
    // VDST = per-lane LDS byte address, VADDR = per-lane 64-bit global addr
    asm volatile("global_load_async_to_lds_b128 %0, %1, off"
                 :: "v"(lds_off), "v"(g) : "memory");
}
__device__ __forceinline__ void wait_async0() {
    asm volatile("s_wait_asynccnt 0x0" ::: "memory");
}
__device__ __forceinline__ unsigned lds_addr(const void* p) {
    return (unsigned)(uintptr_t)p;   // generic LDS ptr: low 32 bits = LDS offset
}

// ---------------------------------------------------------------------------
// Conversion kernels
// ---------------------------------------------------------------------------
__global__ void f2bf4_kernel(const float4* __restrict__ src,
                             uint2* __restrict__ dst, int n4) {
    for (int i = blockIdx.x * blockDim.x + threadIdx.x; i < n4;
         i += gridDim.x * blockDim.x) {
        float4 v = src[i];
        uint2 o;
        o.x = (unsigned)f2bf_bits(v.x) | ((unsigned)f2bf_bits(v.y) << 16);
        o.y = (unsigned)f2bf_bits(v.z) | ((unsigned)f2bf_bits(v.w) << 16);
        dst[i] = o;
    }
}

// wq/wk/wv: [H,E,D] fp32 -> [H,D,E] bf16 (so B-fragments are K-contiguous)
__global__ void wtrans_kernel(const float* __restrict__ src,
                              bf16_t* __restrict__ dst) {
    const int n = HH * EE * DD;
    for (int i = blockIdx.x * blockDim.x + threadIdx.x; i < n;
         i += gridDim.x * blockDim.x) {
        int h = i >> 16;             // / (E*D)
        int rem = i & 0xFFFF;
        int e = rem >> 6;            // / D
        int d = rem & 63;
        dst[((size_t)h * DD + d) * EE + e] = f2bf(src[i]);
    }
}

// ---------------------------------------------------------------------------
// QKV projection: one wave computes a 32(T) x 64(D) tile for one head.
// z=0 -> Q [B,H,T,D], z=1 -> K [B,H,T,D], z=2 -> V transposed [B,H,D,T]
// ---------------------------------------------------------------------------
__global__ __launch_bounds__(32) void qkv_gemm_kernel(
    const bf16_t* __restrict__ xb,
    const bf16_t* __restrict__ wqt,
    const bf16_t* __restrict__ wkt,
    const bf16_t* __restrict__ wvt,
    bf16_t* __restrict__ qo,
    bf16_t* __restrict__ ko,
    bf16_t* __restrict__ vto)
{
    const int lane = threadIdx.x;
    const int m16 = lane & 15, half = lane >> 4;
    const int bt0 = blockIdx.x * 32;
    const int h   = blockIdx.y;
    const int z   = blockIdx.z;
    const int b   = bt0 >> 11;
    const int t0  = bt0 & (TT - 1);

    const bf16_t* wt = (z == 0) ? wqt : (z == 1) ? wkt : wvt;

    v8f acc[2][4] = {};
    const bf16_t* arow[2];
#pragma unroll
    for (int mt = 0; mt < 2; ++mt)
        arow[mt] = xb + (size_t)(bt0 + mt * 16 + m16) * EE;
    const bf16_t* bcol[4];
#pragma unroll
    for (int nt = 0; nt < 4; ++nt)
        bcol[nt] = wt + ((size_t)h * DD + nt * 16 + m16) * EE;

    v16bf a0[2], b0[4], a1[2], b1[4];

#define QKV_LOAD(abuf, bbuf, kk)                                         \
    {                                                                    \
        _Pragma("unroll") for (int mt = 0; mt < 2; ++mt)                 \
            abuf[mt] = load_a(arow[mt], (kk), half);                     \
        _Pragma("unroll") for (int nt = 0; nt < 4; ++nt)                 \
            bbuf[nt] = load_b(bcol[nt], (kk), half);                     \
    }
#define QKV_MMA(abuf, bbuf)                                              \
    {                                                                    \
        _Pragma("unroll") for (int mt = 0; mt < 2; ++mt)                 \
            _Pragma("unroll") for (int nt = 0; nt < 4; ++nt)             \
                acc[mt][nt] = wmma_bf16(abuf[mt], bbuf[nt], acc[mt][nt]);\
    }

    QKV_LOAD(a0, b0, 0);
    for (int k0 = 0; k0 < EE; k0 += 64) {
        QKV_LOAD(a1, b1, k0 + 32);
        __builtin_prefetch(arow[0] + k0 + 128, 0, 1);
        __builtin_prefetch(arow[1] + k0 + 128, 0, 1);
        QKV_MMA(a0, b0);
        if (k0 + 64 < EE) QKV_LOAD(a0, b0, k0 + 64);
        QKV_MMA(a1, b1);
    }
#undef QKV_LOAD
#undef QKV_MMA

    const size_t bh = (size_t)b * HH + h;
    if (z == 2) {
        // transposed V store: consecutive r -> consecutive t -> pack b128
#pragma unroll
        for (int mt = 0; mt < 2; ++mt)
#pragma unroll
            for (int nt = 0; nt < 4; ++nt) {
                union { uint4 u; unsigned short s[8]; } pk;
#pragma unroll
                for (int r = 0; r < 8; ++r)
                    pk.s[r] = f2bf_bits(acc[mt][nt][r]);
                int t = t0 + mt * 16 + 8 * half;
                int d = nt * 16 + m16;
                *(uint4*)(vto + (bh * DD + d) * TT + t) = pk.u;
            }
    } else {
        bf16_t* dst = (z == 0) ? qo : ko;
#pragma unroll
        for (int mt = 0; mt < 2; ++mt)
#pragma unroll
            for (int nt = 0; nt < 4; ++nt)
#pragma unroll
                for (int r = 0; r < 8; ++r) {
                    int t = t0 + mt * 16 + r + 8 * half;
                    int d = nt * 16 + m16;
                    dst[(bh * TT + t) * DD + d] = f2bf(acc[mt][nt][r]);
                }
    }
}

// ---------------------------------------------------------------------------
// Flash attention (causal), 4 waves / block = 64 query rows of one (b,h).
// K/V tiles (32 keys) staged into LDS with async copies, double buffered:
//   issue async loads for s0+32 -> compute block s0 from LDS ->
//   s_wait_asynccnt 0 -> barrier -> swap.
// ---------------------------------------------------------------------------
__global__ __launch_bounds__(128) void attn_kernel(
    const bf16_t* __restrict__ Q,   // [B,H,T,D]
    const bf16_t* __restrict__ K,   // [B,H,T,D]
    const bf16_t* __restrict__ Vt,  // [B,H,D,T]
    bf16_t* __restrict__ O)         // [B,T,H*D]
{
    __shared__ bf16_t Ks[2][32 * DD];   // [s_local][d]   4 KB per buffer
    __shared__ bf16_t Vs[2][DD * 32];   // [d][s_local]   4 KB per buffer
    __shared__ bf16_t Pt[4][16 * 32];   // per-wave P transpose scratch

    const int tid  = threadIdx.x;
    const int w    = tid >> 5;
    const int lane = tid & 31;
    const int m16 = lane & 15, half = lane >> 4;
    const int t0blk = blockIdx.x * 64;
    const int t0    = t0blk + w * 16;      // this wave's 16 query rows
    const int h  = blockIdx.y;
    const int b  = blockIdx.z;
    const size_t bh = (size_t)b * HH + h;

    const bf16_t* kbase = K + bh * TT * DD;
    const bf16_t* vbase = Vt + bh * DD * TT;

    // Q A-fragments for both 32-wide K-steps over D=64
    const bf16_t* qrow = Q + (bh * TT + t0 + m16) * DD;
    v16bf qa0 = load_a(qrow, 0, half);
    v16bf qa1 = load_a(qrow, 32, half);

    float mrun[8], lrun[8];
#pragma unroll
    for (int r = 0; r < 8; ++r) { mrun[r] = -INFINITY; lrun[r] = 0.f; }
    v8f oacc[4] = {};

    const float LOG2E = 1.44269504088896f;
    const float scale = 0.125f;     // D^-0.5

    // Issue async copies of one 32-key K tile + V tile into LDS buffer `buf`.
    // 128 threads x 4 x 16B = 8 KB per call.
    auto issue_tiles = [&](int s0, int buf) {
#pragma unroll
        for (int i = 0; i < 2; ++i) {           // K: 32 rows x 128 B
            int c   = tid + i * 128;            // chunk 0..255 (16B units)
            int row = c >> 3, col = (c & 7) * 8;
            async_ld16(lds_addr(&Ks[buf][row * DD + col]),
                       kbase + (size_t)(s0 + row) * DD + col);
        }
#pragma unroll
        for (int i = 0; i < 2; ++i) {           // V^T: 64 rows x 64 B
            int c = tid + i * 128;
            int d = c >> 2, sc = (c & 3) * 8;
            async_ld16(lds_addr(&Vs[buf][d * 32 + sc]),
                       vbase + (size_t)d * TT + s0 + sc);
        }
    };

    const int s_last = t0blk + 32;   // last 32-key block any wave needs
    issue_tiles(0, 0);
    wait_async0();
    __syncthreads();

    int cur = 0;
    for (int s0 = 0; s0 <= s_last; s0 += 32) {
        if (s0 + 32 <= s_last) issue_tiles(s0 + 32, cur ^ 1);

        if (s0 <= t0 + 15) {   // causal: this wave needs this key block
            // ---- scores from LDS K tile ----
            v8f sc[2] = {};
#pragma unroll
            for (int j = 0; j < 2; ++j) {
                const bf16_t* krow = &Ks[cur][(j * 16 + m16) * DD];
                v16bf kb0 = load_b(krow, 0, half);
                v16bf kb1 = load_b(krow, 32, half);
                sc[j] = wmma_bf16(qa0, kb0, sc[j]);
                sc[j] = wmma_bf16(qa1, kb1, sc[j]);
            }
            // ---- V fragments from LDS, early (overlap softmax) ----
            v16bf vb[4];
#pragma unroll
            for (int nt = 0; nt < 4; ++nt) {
                const bf16_t* vrow = &Vs[cur][(nt * 16 + m16) * 32];
                vb[nt] = load_frag(vrow + half * 8, vrow + 16 + half * 8);
            }

            // ---- scale + causal mask + row stats ----
            float p[2][8], rmax[8], alpha[8], rsum[8];
#pragma unroll
            for (int r = 0; r < 8; ++r) {
                const int trow = t0 + r + 8 * half;
#pragma unroll
                for (int j = 0; j < 2; ++j) {
                    const int s = s0 + j * 16 + m16;
                    float v = sc[j][r] * scale;
                    p[j][r] = (s <= trow) ? v : -INFINITY;
                }
                rmax[r] = fmaxf(p[0][r], p[1][r]);
            }
#pragma unroll
            for (int m = 1; m < 16; m <<= 1)
#pragma unroll
                for (int r = 0; r < 8; ++r)
                    rmax[r] = fmaxf(rmax[r], __shfl_xor(rmax[r], m));

            __threadfence_block();   // order vs previous iteration's P reads
#pragma unroll
            for (int r = 0; r < 8; ++r) {
                float mnew = fmaxf(mrun[r], rmax[r]);
                alpha[r] = exp2f((mrun[r] - mnew) * LOG2E);
                mrun[r] = mnew;
                float sum = 0.f;
#pragma unroll
                for (int j = 0; j < 2; ++j) {
                    float e = exp2f((p[j][r] - mnew) * LOG2E);
                    sum += e;
                    Pt[w][(r + 8 * half) * 32 + j * 16 + m16] = f2bf(e);
                }
                rsum[r] = sum;
            }
#pragma unroll
            for (int m = 1; m < 16; m <<= 1)
#pragma unroll
                for (int r = 0; r < 8; ++r)
                    rsum[r] += __shfl_xor(rsum[r], m);
#pragma unroll
            for (int r = 0; r < 8; ++r)
                lrun[r] = lrun[r] * alpha[r] + rsum[r];
#pragma unroll
            for (int nt = 0; nt < 4; ++nt)
#pragma unroll
                for (int r = 0; r < 8; ++r)
                    oacc[nt][r] *= alpha[r];

            __threadfence_block();   // P stores visible to all lanes (per-wave)

            // ---- PV: A = P(16x32) from per-wave LDS scratch ----
            const bf16_t* prow = &Pt[w][m16 * 32];
            v16bf pa = load_frag(prow + half * 8, prow + 16 + half * 8);
#pragma unroll
            for (int nt = 0; nt < 4; ++nt)
                oacc[nt] = wmma_bf16(pa, vb[nt], oacc[nt]);
        }

        wait_async0();     // this wave's async copies into buffer cur^1 done
        __syncthreads();   // all waves' copies done; readers done with cur
        cur ^= 1;
    }

    // ---- normalize and emit bf16 [B,T,H*D] ----
    float inv[8];
#pragma unroll
    for (int r = 0; r < 8; ++r) inv[r] = 1.0f / lrun[r];
#pragma unroll
    for (int nt = 0; nt < 4; ++nt)
#pragma unroll
        for (int r = 0; r < 8; ++r) {
            int t = t0 + r + 8 * half;
            int col = h * DD + nt * 16 + m16;
            O[((size_t)b * TT + t) * EE + col] = f2bf(oacc[nt][r] * inv[r]);
        }
}

// ---------------------------------------------------------------------------
// Output projection: out = Ob @ wo^T + bo.  32x64 tile per wave, ping-pong.
// ---------------------------------------------------------------------------
__global__ __launch_bounds__(32) void proj_kernel(
    const bf16_t* __restrict__ Ob,   // [B*T, E]
    const bf16_t* __restrict__ Wob,  // [E, E] bf16
    const float*  __restrict__ bo,   // [E]
    float* __restrict__ out)         // [B*T, E]
{
    const int lane = threadIdx.x;
    const int m16 = lane & 15, half = lane >> 4;
    const int bt0 = blockIdx.x * 32;
    const int n0  = blockIdx.y * 64;

    v8f acc[2][4] = {};
    const bf16_t* arow[2];
#pragma unroll
    for (int mt = 0; mt < 2; ++mt)
        arow[mt] = Ob + (size_t)(bt0 + mt * 16 + m16) * EE;
    const bf16_t* bcol[4];
#pragma unroll
    for (int nt = 0; nt < 4; ++nt)
        bcol[nt] = Wob + (size_t)(n0 + nt * 16 + m16) * EE;

    v16bf a0[2], b0[4], a1[2], b1[4];

#define PRJ_LOAD(abuf, bbuf, kk)                                         \
    {                                                                    \
        _Pragma("unroll") for (int mt = 0; mt < 2; ++mt)                 \
            abuf[mt] = load_a(arow[mt], (kk), half);                     \
        _Pragma("unroll") for (int nt = 0; nt < 4; ++nt)                 \
            bbuf[nt] = load_b(bcol[nt], (kk), half);                     \
    }
#define PRJ_MMA(abuf, bbuf)                                              \
    {                                                                    \
        _Pragma("unroll") for (int mt = 0; mt < 2; ++mt)                 \
            _Pragma("unroll") for (int nt = 0; nt < 4; ++nt)             \
                acc[mt][nt] = wmma_bf16(abuf[mt], bbuf[nt], acc[mt][nt]);\
    }

    PRJ_LOAD(a0, b0, 0);
    for (int k0 = 0; k0 < EE; k0 += 64) {
        PRJ_LOAD(a1, b1, k0 + 32);
        __builtin_prefetch(arow[0] + k0 + 128, 0, 1);
        __builtin_prefetch(arow[1] + k0 + 128, 0, 1);
        PRJ_MMA(a0, b0);
        if (k0 + 64 < EE) PRJ_LOAD(a0, b0, k0 + 64);
        PRJ_MMA(a1, b1);
    }
#undef PRJ_LOAD
#undef PRJ_MMA

#pragma unroll
    for (int nt = 0; nt < 4; ++nt) {
        const int n = n0 + nt * 16 + m16;
        const float bias = bo[n];
#pragma unroll
        for (int mt = 0; mt < 2; ++mt)
#pragma unroll
            for (int r = 0; r < 8; ++r) {
                int bt = bt0 + mt * 16 + r + 8 * half;
                out[(size_t)bt * EE + n] = acc[mt][nt][r] + bias;
            }
    }
}

// ---------------------------------------------------------------------------
// Host launcher
// ---------------------------------------------------------------------------
extern "C" void kernel_launch(void* const* d_in, const int* in_sizes, int n_in,
                              void* d_out, int out_size, void* d_ws, size_t ws_size,
                              hipStream_t stream) {
    (void)in_sizes; (void)n_in; (void)out_size; (void)ws_size;
    const float* x  = (const float*)d_in[0];
    const float* wq = (const float*)d_in[1];
    const float* wk = (const float*)d_in[2];
    const float* wv = (const float*)d_in[3];
    const float* wo = (const float*)d_in[4];
    const float* bo = (const float*)d_in[5];
    float* out = (float*)d_out;

    char* ws = (char*)d_ws;
    size_t off = 0;
    bf16_t* xb  = (bf16_t*)(ws + off); off += (size_t)BB * TT * EE * 2;   // 16 MB
    bf16_t* wqt = (bf16_t*)(ws + off); off += (size_t)HH * DD * EE * 2;   //  2 MB
    bf16_t* wkt = (bf16_t*)(ws + off); off += (size_t)HH * DD * EE * 2;
    bf16_t* wvt = (bf16_t*)(ws + off); off += (size_t)HH * DD * EE * 2;
    bf16_t* wob = (bf16_t*)(ws + off); off += (size_t)EE * EE * 2;
    bf16_t* qb  = (bf16_t*)(ws + off); off += (size_t)BB * HH * TT * DD * 2; // 16 MB
    bf16_t* kb  = (bf16_t*)(ws + off); off += (size_t)BB * HH * TT * DD * 2;
    bf16_t* vtb = (bf16_t*)(ws + off); off += (size_t)BB * HH * TT * DD * 2;
    bf16_t* ob  = (bf16_t*)(ws + off); off += (size_t)BB * TT * EE * 2;   // ~88 MB

    // precision conversion / weight transpose
    f2bf4_kernel<<<1024, 256, 0, stream>>>((const float4*)x, (uint2*)xb,
                                           BB * TT * EE / 4);
    wtrans_kernel<<<1024, 256, 0, stream>>>(wq, wqt);
    wtrans_kernel<<<1024, 256, 0, stream>>>(wk, wkt);
    wtrans_kernel<<<1024, 256, 0, stream>>>(wv, wvt);
    f2bf4_kernel<<<256, 256, 0, stream>>>((const float4*)wo, (uint2*)wob,
                                          EE * EE / 4);

    // QKV projection (z selects q/k/v)
    qkv_gemm_kernel<<<dim3(BB * TT / 32, HH, 3), 32, 0, stream>>>(
        xb, wqt, wkt, wvt, qb, kb, vtb);

    // flash attention (4 waves/block, async K/V staging)
    attn_kernel<<<dim3(TT / 64, HH, BB), 128, 0, stream>>>(qb, kb, vtb, ob);

    // output projection + bias
    proj_kernel<<<dim3(BB * TT / 32, EE / 64), 32, 0, stream>>>(ob, wob, bo, out);
}
